// VQVAE_11682311045528
// MI455X (gfx1250) — compile-verified, matrix operand order checked by reading, not run on hardware
//
#include <hip/hip_runtime.h>

// ---------------------------------------------------------------------------
// MI455X (gfx1250) VQ-VAE forward.
// All dense math (convs as implicit GEMM, attention GEMMs) goes through
// V_WMMA_F32_16X16X32_F16 (wave32). A 4-wave workgroup owns a 64x64 output
// tile: each wave stages one 16-row A tile and one 16-col B tile into LDS
// (fragment order), then computes one 32x32 quadrant (2x2 WMMAs). Staged
// data is reused by all four waves -> 8 lane-loads per WMMA, which is what
// the WMMA-rate / L2-bandwidth balance of this chip demands.
//
// Instruction-mix features:
//  * LDS staged directly in WMMA fragment order -> ds_load_b128 frag reads
//  * unconditional clamped global loads + v_cndmask (no exec diamonds)
//  * GEMM has zero guards (all dims multiples of 64 in this network)
//  * log2 spatial dims (powers of two) -> shifts instead of div/mod
//  * int32 addressing (all tensors < 2^31 elements)
//  * batched, vectorizable epilogue (8 consecutive floats per lane in conv)
// ---------------------------------------------------------------------------

typedef _Float16 v16h __attribute__((ext_vector_type(16)));
typedef float    v8f  __attribute__((ext_vector_type(8)));

// Inverse of the ISA 7.12.2 16-bit fragment layout: LDS slot (lane*16+e) that
// holds element (rowlane = M or N within 16, kk = K within the 32-chunk).
__device__ __forceinline__ int frag_slot(int rowlane, int kk) {
    int base = kk & 16;          // 0 or 16
    int kr   = kk & 15;
    int hi   = kr >> 3;          // which half-wave holds it
    int krr  = kr & 7;
    int e    = (((krr >> 1) + (base ? 4 : 0)) << 1) + (krr & 1);
    return ((rowlane + (hi << 4)) << 4) + e;
}

__device__ __forceinline__ int imin(int a, int b) { return a < b ? a : b; }
__device__ __forceinline__ int imax(int a, int b) { return a > b ? a : b; }

// ---------------------------------------------------------------------------
// Implicit-GEMM convolution (and transposed conv via parity gather).
// 4-wave block, 64x64 tile; wave w stages A m-tile w and B n-tile w,
// computes quadrant (mi = w&1, ni = w>>1). H = 1<<lH, OH = 1<<lOH,
// P = OH*OH is a multiple of 64 here (>= 1024).
// ---------------------------------------------------------------------------
template <int KS, int STRIDE, bool TR>
__global__ void __launch_bounds__(128)
conv_wmma_kernel(const float* __restrict__ X, const float* __restrict__ Wt,
                 const float* __restrict__ bias, const float* __restrict__ resid,
                 float* __restrict__ Y,
                 int Cin, int Cout, int lH, int lOH, int pad)
{
    constexpr int KK = KS * KS;
    const int lane = threadIdx.x & 31;
    const int w    = threadIdx.x >> 5;        // wave id 0..3
    const int n    = blockIdx.z;
    const int p0   = blockIdx.x * 64;
    const int co0  = blockIdx.y * 64;
    const int H    = 1 << lH;
    const int OH   = 1 << lOH;
    const int P    = 1 << (2 * lOH);
    const int Ktot = Cin * KK;

    __shared__ alignas(32) _Float16 As[2048];  // 4 fragment-order A tiles
    __shared__ alignas(32) _Float16 Bs[2048];  // 4 fragment-order B tiles

    v8f acc[2][2] = {};                        // [m-subtile][n-subtile]

    const int mrow = lane & 15;
    const int kkb  = lane & 16;
    const int mi   = w & 1;                    // quadrant row
    const int ni   = w >> 1;                   // quadrant col

    for (int k0 = 0; k0 < Ktot; k0 += 32) {
        // ---- stage A (im2col gather, f32 -> f16): wave w -> m-tile w ----
        {
            int op = p0 + w * 16 + mrow;
            int oh = op >> lOH, ow = op & (OH - 1);
            float av[16]; bool avd[16];
            #pragma unroll
            for (int t = 0; t < 16; ++t) {
                int k  = k0 + kkb + t;
                int kc = imin(k, Ktot - 1);
                int ci = kc / KK, r = kc - ci * KK;
                int kh = r / KS,  kw = r - kh * KS;
                int ih, iw;
                bool vs;
                if (!TR) {
                    ih = oh * STRIDE - pad + kh;
                    iw = ow * STRIDE - pad + kw;
                    vs = ((unsigned)ih < (unsigned)H) & ((unsigned)iw < (unsigned)H);
                } else {
                    int th = oh + pad - kh;
                    int tw = ow + pad - kw;
                    vs = (th >= 0) & (tw >= 0) &
                         ((th % STRIDE) == 0) & ((tw % STRIDE) == 0);
                    ih = th / STRIDE;
                    iw = tw / STRIDE;
                    vs = vs & (ih < H) & (iw < H);
                }
                int ihc = imin(imax(ih, 0), H - 1);
                int iwc = imin(imax(iw, 0), H - 1);
                int off = ((n * Cin + ci) << (2 * lH)) + (ihc << lH) + iwc;
                av[t]  = X[off];
                avd[t] = (k < Ktot) & vs;
            }
            _Float16* dst = As + w * 512;
            #pragma unroll
            for (int t = 0; t < 16; ++t)
                dst[frag_slot(mrow, kkb + t)] = (_Float16)(avd[t] ? av[t] : 0.f);
        }

        // ---- stage B (weights, f32 -> f16): wave w -> n-tile w ----
        {
            int kB  = k0 + lane;
            int kcB = imin(kB, Ktot - 1);
            bool vk = kB < Ktot;
            int wbase;
            if (!TR) {
                wbase = kcB;                       // + co*Ktot per column
            } else {
                int ci = kcB / KK, rr = kcB - ci * KK;
                wbase = ci * Cout * KK + rr;       // + co*KK per column
            }
            float bv[16]; bool bvd[16];
            #pragma unroll
            for (int t = 0; t < 16; ++t) {
                int co  = co0 + w * 16 + t;
                int coc = imin(co, Cout - 1);
                int woff = TR ? (wbase + coc * KK) : (coc * Ktot + wbase);
                bv[t]  = Wt[woff];
                bvd[t] = vk & (co < Cout);
            }
            _Float16* dst = Bs + w * 512;
            #pragma unroll
            for (int t = 0; t < 16; ++t)
                dst[frag_slot(t, lane)] = (_Float16)(bvd[t] ? bv[t] : 0.f);
        }

        if (!TR && (k0 + 32 < Ktot))
            __builtin_prefetch(Wt + (co0 + w * 16) * Ktot + k0 + 32 + lane, 0, 1);

        __syncthreads();
        v16h a0 = ((const v16h*)(As + (2 * mi + 0) * 512))[lane];
        v16h a1 = ((const v16h*)(As + (2 * mi + 1) * 512))[lane];
        v16h b0 = ((const v16h*)(Bs + (2 * ni + 0) * 512))[lane];
        v16h b1 = ((const v16h*)(Bs + (2 * ni + 1) * 512))[lane];
        acc[0][0] = __builtin_amdgcn_wmma_f32_16x16x32_f16(
            false, a0, false, b0, (short)0, acc[0][0], false, false);
        acc[0][1] = __builtin_amdgcn_wmma_f32_16x16x32_f16(
            false, a0, false, b1, (short)0, acc[0][1], false, false);
        acc[1][0] = __builtin_amdgcn_wmma_f32_16x16x32_f16(
            false, a1, false, b0, (short)0, acc[1][0], false, false);
        acc[1][1] = __builtin_amdgcn_wmma_f32_16x16x32_f16(
            false, a1, false, b1, (short)0, acc[1][1], false, false);
        __syncthreads();
    }

    // ---- epilogue: bias + optional residual; 8 consecutive pixels per lane
    const int nn  = lane & 15;
    const int hi8 = (lane >> 4) << 3;
    #pragma unroll
    for (int cn = 0; cn < 2; ++cn) {
        int co = co0 + ni * 32 + cn * 16 + nn;
        if (co < Cout) {
            float bvs = bias[co];
            #pragma unroll
            for (int pm = 0; pm < 2; ++pm) {
                int base = (n * Cout + co) * P + p0 + mi * 32 + pm * 16 + hi8;
                if (resid) {
                    float rv[8];
                    #pragma unroll
                    for (int r = 0; r < 8; ++r) rv[r] = resid[base + r];
                    #pragma unroll
                    for (int r = 0; r < 8; ++r)
                        Y[base + r] = acc[pm][cn][r] + bvs + rv[r];
                } else {
                    #pragma unroll
                    for (int r = 0; r < 8; ++r)
                        Y[base + r] = acc[pm][cn][r] + bvs;
                }
            }
        }
    }
}

// ---------------------------------------------------------------------------
// Strided / batched WMMA GEMM:  Y = alpha * A*B + bias + resid
// 4-wave block, 64x64 tile. ASSUMES M % 64 == 0, N % 64 == 0, K % 32 == 0
// (true for every call here) -> zero guards. z batch: (z/zdiv, z%zdiv).
// ---------------------------------------------------------------------------
__global__ void __launch_bounds__(128)
gemm_wmma_kernel(const float* __restrict__ A, const float* __restrict__ B,
                 const float* __restrict__ bias, const float* __restrict__ resid,
                 float* __restrict__ Y, int M, int Nn, int K, int zdiv,
                 int a1, int a2, int sAm, int sAk,
                 int b1, int b2, int sBk, int sBn,
                 int y1, int y2, int sYm, int sYn,
                 float alpha)
{
    const int lane = threadIdx.x & 31;
    const int w    = threadIdx.x >> 5;
    const int z = blockIdx.z, zb = z / zdiv, zh = z - zb * zdiv;
    const float* Ab = A + zb * a1 + zh * a2;
    const float* Bb = B + zb * b1 + zh * b2;
    const int yoff0 = zb * y1 + zh * y2;
    const int m0 = blockIdx.x * 64, n0 = blockIdx.y * 64;

    __shared__ alignas(32) _Float16 As[2048];
    __shared__ alignas(32) _Float16 Bs[2048];

    v8f acc[2][2] = {};

    const int mrow = lane & 15;
    const int kkb  = lane & 16;
    const int mi   = w & 1;
    const int ni   = w >> 1;

    for (int k0 = 0; k0 < K; k0 += 32) {
        {
            const float* arow = Ab + (m0 + w * 16 + mrow) * sAm;
            float av[16];
            #pragma unroll
            for (int t = 0; t < 16; ++t)
                av[t] = arow[(k0 + kkb + t) * sAk];
            _Float16* dst = As + w * 512;
            #pragma unroll
            for (int t = 0; t < 16; ++t)
                dst[frag_slot(mrow, kkb + t)] = (_Float16)av[t];
        }
        {
            const float* brow = Bb + (k0 + lane) * sBk;
            float bv[16];
            #pragma unroll
            for (int t = 0; t < 16; ++t)
                bv[t] = brow[(n0 + w * 16 + t) * sBn];
            _Float16* dst = Bs + w * 512;
            #pragma unroll
            for (int t = 0; t < 16; ++t)
                dst[frag_slot(t, lane)] = (_Float16)bv[t];
        }
        __syncthreads();
        v16h a0 = ((const v16h*)(As + (2 * mi + 0) * 512))[lane];
        v16h a1 = ((const v16h*)(As + (2 * mi + 1) * 512))[lane];
        v16h b0 = ((const v16h*)(Bs + (2 * ni + 0) * 512))[lane];
        v16h b1 = ((const v16h*)(Bs + (2 * ni + 1) * 512))[lane];
        acc[0][0] = __builtin_amdgcn_wmma_f32_16x16x32_f16(
            false, a0, false, b0, (short)0, acc[0][0], false, false);
        acc[0][1] = __builtin_amdgcn_wmma_f32_16x16x32_f16(
            false, a0, false, b1, (short)0, acc[0][1], false, false);
        acc[1][0] = __builtin_amdgcn_wmma_f32_16x16x32_f16(
            false, a1, false, b0, (short)0, acc[1][0], false, false);
        acc[1][1] = __builtin_amdgcn_wmma_f32_16x16x32_f16(
            false, a1, false, b1, (short)0, acc[1][1], false, false);
        __syncthreads();
    }

    const int nn  = lane & 15;
    const int hi8 = (lane >> 4) << 3;
    #pragma unroll
    for (int cn = 0; cn < 2; ++cn) {
        int nidx = n0 + ni * 32 + cn * 16 + nn;
        float bvs = bias ? bias[nidx] : 0.f;
        #pragma unroll
        for (int pm = 0; pm < 2; ++pm) {
            int base = yoff0 + (m0 + mi * 32 + pm * 16 + hi8) * sYm + nidx * sYn;
            if (resid) {
                float rv[8];
                #pragma unroll
                for (int r = 0; r < 8; ++r) rv[r] = resid[base + r * sYm];
                #pragma unroll
                for (int r = 0; r < 8; ++r)
                    Y[base + r * sYm] = alpha * acc[pm][cn][r] + bvs + rv[r];
            } else {
                #pragma unroll
                for (int r = 0; r < 8; ++r)
                    Y[base + r * sYm] = alpha * acc[pm][cn][r] + bvs;
            }
        }
    }
}

// ---------------------------------------------------------------------------
// GroupNorm: per-(batch,group) stats, then fused affine + optional SiLU.
// ---------------------------------------------------------------------------
__global__ void __launch_bounds__(256)
gn_stats_kernel(const float* __restrict__ x, float* __restrict__ mean,
                float* __restrict__ rstd, int C, int HW, int G)
{
    int bg = blockIdx.x;
    int b = bg / G, g = bg - b * G;
    int cpg = C / G;
    int n = cpg * HW;
    int base = (b * C + g * cpg) * HW;
    float s = 0.f, ss = 0.f;
    for (int i = threadIdx.x; i < n; i += 256) {
        float v = x[base + i];
        s += v; ss += v * v;
    }
    __shared__ float r1[256], r2[256];
    r1[threadIdx.x] = s; r2[threadIdx.x] = ss;
    __syncthreads();
    for (int st = 128; st > 0; st >>= 1) {
        if (threadIdx.x < st) {
            r1[threadIdx.x] += r1[threadIdx.x + st];
            r2[threadIdx.x] += r2[threadIdx.x + st];
        }
        __syncthreads();
    }
    if (threadIdx.x == 0) {
        float m = r1[0] / (float)n;
        float v = r2[0] / (float)n - m * m;
        mean[bg] = m;
        rstd[bg] = rsqrtf(v + 1e-5f);
    }
}

__global__ void __launch_bounds__(256)
gn_apply_kernel(const float* __restrict__ x, float* __restrict__ y,
                const float* __restrict__ mean, const float* __restrict__ rstd,
                const float* __restrict__ gamma, const float* __restrict__ beta,
                int C, int HW, int G, int do_silu)
{
    int idx = blockIdx.x * 256 + threadIdx.x;
    int tot = 4 * C * HW;
    if (idx >= tot) return;
    int b = idx / (C * HW);
    int c = (idx / HW) % C;
    int cpg = C / G;
    int bg = b * G + c / cpg;
    float v = (x[idx] - mean[bg]) * rstd[bg] * gamma[c] + beta[c];
    if (do_silu) v = v / (1.f + __expf(-v));
    y[idx] = v;
}

// ---------------------------------------------------------------------------
// Row softmax (rows = B*heads*L, row length L).
// ---------------------------------------------------------------------------
__global__ void __launch_bounds__(256)
softmax_rows_kernel(float* __restrict__ S, int L)
{
    float* p = S + (long long)blockIdx.x * L;
    __shared__ float red[256];
    int tid = threadIdx.x;

    float mx = -3.0e38f;
    for (int i = tid; i < L; i += 256) mx = fmaxf(mx, p[i]);
    red[tid] = mx; __syncthreads();
    for (int st = 128; st > 0; st >>= 1) {
        if (tid < st) red[tid] = fmaxf(red[tid], red[tid + st]);
        __syncthreads();
    }
    mx = red[0];
    __syncthreads();

    float sum = 0.f;
    for (int i = tid; i < L; i += 256) {
        float e = __expf(p[i] - mx);
        p[i] = e; sum += e;
    }
    red[tid] = sum; __syncthreads();
    for (int st = 128; st > 0; st >>= 1) {
        if (tid < st) red[tid] += red[tid + st];
        __syncthreads();
    }
    float inv = 1.f / red[0];
    for (int i = tid; i < L; i += 256) p[i] *= inv;
}

// ---------------------------------------------------------------------------
// Vector quantization: argmin over 8192 codes (C=3) + gather. NCHW layout.
// ---------------------------------------------------------------------------
__global__ void __launch_bounds__(256)
vq_kernel(const float* __restrict__ z, const float* __restrict__ emb,
          float* __restrict__ q, int L, int Kc)
{
    int i = blockIdx.x * 256 + threadIdx.x;
    if (i >= 4 * L) return;
    int b = i / L, l = i - b * L;
    float z0 = z[(b * 3 + 0) * L + l];
    float z1 = z[(b * 3 + 1) * L + l];
    float z2 = z[(b * 3 + 2) * L + l];
    float best = 3.4e38f; int bi = 0;
    #pragma unroll 4
    for (int k = 0; k < Kc; ++k) {
        float d0 = z0 - emb[k * 3 + 0];
        float d1 = z1 - emb[k * 3 + 1];
        float d2 = z2 - emb[k * 3 + 2];
        float d = d0 * d0 + d1 * d1 + d2 * d2;
        if (d < best) { best = d; bi = k; }
    }
    q[(b * 3 + 0) * L + l] = emb[bi * 3 + 0];
    q[(b * 3 + 1) * L + l] = emb[bi * 3 + 1];
    q[(b * 3 + 2) * L + l] = emb[bi * 3 + 2];
}

__global__ void __launch_bounds__(256)
vq_loss_kernel(const float* __restrict__ z, const float* __restrict__ q,
               float* __restrict__ cb, float* __restrict__ cm, int n)
{
    __shared__ float red[256];
    float s = 0.f;
    for (int i = threadIdx.x; i < n; i += 256) {
        float d = q[i] - z[i];
        s += d * d;
    }
    red[threadIdx.x] = s; __syncthreads();
    for (int st = 128; st > 0; st >>= 1) {
        if (threadIdx.x < st) red[threadIdx.x] += red[threadIdx.x + st];
        __syncthreads();
    }
    if (threadIdx.x == 0) {
        float m = red[0] / (float)n;
        *cb = m;  // codebook and commitment losses are numerically identical
        *cm = m;
    }
}

// ---------------------------------------------------------------------------
// Host-side parameter bookkeeping (JAX pytree flatten order: dict keys sorted)
// ---------------------------------------------------------------------------
struct ConvP { const float* b; const float* w; };
struct GnP   { const float* b; const float* g; };
struct ResP  { ConvP c1, c2, res; GnP g1, g2; };
struct AttnP { GnP gn; const float* in_b; const float* in_w;
               const float* out_b; const float* out_w; };
struct MidP  { AttnP at[2]; ResP rs[3]; };
struct DownP { ConvP down; ResP r0, r1; };
struct UpP   { ResP r0, r1; ConvP up; };

static inline int ilog2(int v) { int l = 0; while ((1 << l) < v) ++l; return l; }

extern "C" void kernel_launch(void* const* d_in, const int* in_sizes, int n_in,
                              void* d_out, int out_size, void* d_ws, size_t ws_size,
                              hipStream_t stream)
{
    (void)in_sizes; (void)n_in; (void)out_size; (void)ws_size;

    const float* x = (const float*)d_in[0];   // [4,3,256,256]
    int ii = 1;
    auto nx = [&]() -> const float* { return (const float*)d_in[ii++]; };

    auto getConv = [&]() { ConvP c; c.b = nx(); c.w = nx(); return c; };
    auto getGn   = [&]() { GnP g; g.b = nx(); g.g = nx(); return g; };
    auto getRes  = [&]() {
        ResP r; r.c1 = getConv(); r.c2 = getConv();
        r.g1 = getGn(); r.g2 = getGn(); r.res = getConv(); return r;
    };
    auto getAttn = [&]() {
        AttnP a; a.gn = getGn(); a.in_b = nx(); a.in_w = nx();
        a.out_b = nx(); a.out_w = nx(); return a;
    };
    auto getMid  = [&]() {
        MidP m; m.at[0] = getAttn(); m.at[1] = getAttn();
        m.rs[0] = getRes(); m.rs[1] = getRes(); m.rs[2] = getRes(); return m;
    };

    // Top-level sorted keys:
    ConvP dec_in_p  = getConv();             // dec_in
    MidP  dec_mid   = getMid();              // dec_mids[0]
    GnP   dec_norm  = getGn();               // dec_norm
    ConvP dec_out_p = getConv();             // dec_out
    DownP downs_[3];                         // downs
    for (int i = 0; i < 3; ++i) {
        downs_[i].down = getConv();          // 'down' < 'res'
        downs_[i].r0 = getRes();
        downs_[i].r1 = getRes();
    }
    const float* embedding = nx();           // embedding
    ConvP enc_in_p  = getConv();             // enc_in
    MidP  enc_mid   = getMid();              // enc_mids[0]
    GnP   enc_norm  = getGn();               // enc_norm
    ConvP enc_out_p = getConv();             // enc_out
    ConvP post_q_p  = getConv();             // post_quant
    ConvP pre_q_p   = getConv();             // pre_quant
    UpP ups_[3];                             // ups
    for (int i = 0; i < 3; ++i) {
        ups_[i].r0 = getRes();               // 'res' < 'up'
        ups_[i].r1 = getRes();
        ups_[i].up = getConv();
    }

    // ---- workspace arena: 4 ping-pong buffers + GN stats ----
    const long long BUFN = 33554432LL;       // 4*128*256*256 floats (max tensor)
    float* arr[4];
    arr[0] = (float*)d_ws;
    arr[1] = arr[0] + BUFN;
    arr[2] = arr[1] + BUFN;
    arr[3] = arr[2] + BUFN;
    float* gmean = arr[3] + BUFN;
    float* grstd = gmean + 1024;

    float* dout  = (float*)d_out;
    float* recon = dout;                     // [4,3,256,256]
    float* zq    = dout + 786432;            // [4,3,32,32]
    float* cbp   = dout + 786432 + 12288;
    float* cmp   = cbp + 1;

    auto others = [&](float* c, float** o) {
        int j = 0;
        for (int i = 0; i < 4; ++i) if (arr[i] != c) o[j++] = arr[i];
    };

    auto convrun = [&](const float* in, const ConvP& p, const float* resid,
                       float* out, int Cin, int Hh, int Cout,
                       int Kk, int strd, int pd, bool tr) {
        int OH = tr ? ((Hh - 1) * strd - 2 * pd + Kk) : ((Hh + 2 * pd - Kk) / strd + 1);
        int lH = ilog2(Hh), lOH = ilog2(OH);
        dim3 g((OH * OH + 63) / 64, (Cout + 63) / 64, 4);
        if (tr)
            conv_wmma_kernel<4, 2, true><<<g, 128, 0, stream>>>(
                in, p.w, p.b, resid, out, Cin, Cout, lH, lOH, pd);
        else if (Kk == 3)
            conv_wmma_kernel<3, 1, false><<<g, 128, 0, stream>>>(
                in, p.w, p.b, resid, out, Cin, Cout, lH, lOH, pd);
        else if (Kk == 1)
            conv_wmma_kernel<1, 1, false><<<g, 128, 0, stream>>>(
                in, p.w, p.b, resid, out, Cin, Cout, lH, lOH, pd);
        else
            conv_wmma_kernel<4, 2, false><<<g, 128, 0, stream>>>(
                in, p.w, p.b, resid, out, Cin, Cout, lH, lOH, pd);
    };

    auto gnrun = [&](const float* in, const GnP& p, float* out, int C,
                     int HW, bool silu) {
        gn_stats_kernel<<<dim3(4 * 32), 256, 0, stream>>>(in, gmean, grstd, C, HW, 32);
        int tot = 4 * C * HW;
        gn_apply_kernel<<<dim3((tot + 255) / 256), 256, 0, stream>>>(
            in, out, gmean, grstd, p.g, p.b, C, HW, 32, silu ? 1 : 0);
    };

    auto gemmrun = [&](const float* Am, const float* Bm, const float* bias,
                       const float* resid, float* Ym, int M, int Nn, int K,
                       int Z, int zdiv,
                       int a1, int a2, int sAm, int sAk,
                       int b1, int b2, int sBk, int sBn,
                       int y1, int y2, int sYm, int sYn,
                       float alpha) {
        dim3 g(M / 64, Nn / 64, Z);
        gemm_wmma_kernel<<<g, 128, 0, stream>>>(Am, Bm, bias, resid, Ym, M, Nn, K,
                                                zdiv, a1, a2, sAm, sAk,
                                                b1, b2, sBk, sBn,
                                                y1, y2, sYm, sYn, alpha);
    };

    auto resnet = [&](const ResP& r, const float* in, float* out, float* t1,
                      float* t2, int Cin, int Cout, int Hh) {
        int HW = Hh * Hh;
        gnrun(in, r.g1, t1, Cin, HW, true);
        convrun(t1, r.c1, nullptr, t2, Cin, Hh, Cout, 3, 1, 1, false);
        gnrun(t2, r.g2, t1, Cout, HW, true);
        convrun(in, r.res, nullptr, out, Cin, Hh, Cout, 1, 1, 0, false);
        convrun(t1, r.c2, out, out, Cout, Hh, Cout, 3, 1, 1, false);
    };

    // attention: input `in` stays live for the residual; result lands in t1.
    auto attention = [&](const AttnP& a, const float* in, float* t1, float* t2,
                         float* obuf, int C, int Hh) {
        int L = Hh * Hh;
        int C3 = 3 * C;
        gnrun(in, a.gn, t1, C, L, false);
        // qkv = t^T @ in_w^T + in_b   -> t2 [B, L, 3C]
        gemmrun(t1, a.in_w, a.in_b, nullptr, t2, L, C3, C, 4, 1,
                C * L, 0, 1, L,
                0, 0, 1, C,
                L * C3, 0, C3, 1, 1.0f);
        // scores = Q @ K^T / sqrt(hd)  -> t1 [B*h, L, L]
        gemmrun(t2, t2 + C, nullptr, nullptr, t1, L, L, 64, 16, 4,
                L * C3, 64, C3, 1,
                L * C3, 64, 1, C3,
                4 * L * L, L * L, L, 1, 0.125f);
        softmax_rows_kernel<<<dim3(16 * L), 256, 0, stream>>>(t1, L);
        // O = att @ V  -> obuf [B, L, C]
        gemmrun(t1, t2 + 2 * C, nullptr, nullptr, obuf, L, 64, L, 16, 4,
                4 * L * L, L * L, L, 1,
                L * C3, 64, C3, 1,
                L * C, 64, C, 1, 1.0f);
        // proj + residual, transposed store to NCHW -> t1 [B, C, L]
        gemmrun(obuf, a.out_w, a.out_b, in, t1, L, C, C, 4, 1,
                L * C, 0, C, 1,
                0, 0, 1, C,
                C * L, 0, 1, L, 1.0f);
    };

    auto midrun = [&](const MidP& m, float* xb, float* u, float* v, float* w_,
                      int C, int Hh) -> float* {
        resnet(m.rs[0], xb, u, v, w_, C, C, Hh);       // -> u
        attention(m.at[0], u, v, w_, xb, C, Hh);       // -> v
        resnet(m.rs[1], v, u, w_, xb, C, C, Hh);       // -> u
        attention(m.at[1], u, v, w_, xb, C, Hh);       // -> v
        resnet(m.rs[2], v, u, w_, xb, C, C, Hh);       // -> u
        return u;
    };

    // ======================= forward pass =======================
    float* o[3];

    // encoder input conv: 3 -> 64 @ 256^2
    convrun(x, enc_in_p, nullptr, arr[0], 3, 256, 64, 3, 1, 1, false);
    float* cur = arr[0];
    int H = 256;
    const int dch[4] = {64, 128, 256, 256};

    for (int i = 0; i < 3; ++i) {
        int cin = dch[i], cout = dch[i + 1];
        others(cur, o);
        resnet(downs_[i].r0, cur, o[0], o[1], o[2], cin, cout, H);
        resnet(downs_[i].r1, o[0], cur, o[1], o[2], cout, cout, H);
        convrun(cur, downs_[i].down, nullptr, o[0], cout, H, cout, 4, 2, 1, false);
        cur = o[0];
        H >>= 1;
    }
    // H == 32, 256 channels

    others(cur, o);
    cur = midrun(enc_mid, cur, o[0], o[1], o[2], 256, 32);

    others(cur, o);
    gnrun(cur, enc_norm, o[0], 256, 1024, true);
    convrun(o[0], enc_out_p, nullptr, o[1], 256, 32, 3, 3, 1, 1, false);
    convrun(o[1], pre_q_p, nullptr, o[2], 3, 32, 3, 1, 1, 0, false);

    // quantize: z in o[2], q (straight-through output) -> d_out z slot
    vq_kernel<<<dim3((4 * 1024 + 255) / 256), 256, 0, stream>>>(o[2], embedding, zq, 1024, 8192);
    vq_loss_kernel<<<dim3(1), 256, 0, stream>>>(o[2], zq, cbp, cmp, 4 * 3 * 1024);

    // decoder
    convrun(zq, post_q_p, nullptr, o[0], 3, 32, 3, 1, 1, 0, false);
    convrun(o[0], dec_in_p, nullptr, o[1], 3, 32, 256, 3, 1, 1, false);
    cur = o[1];

    others(cur, o);
    cur = midrun(dec_mid, cur, o[0], o[1], o[2], 256, 32);

    const int uin[3]  = {256, 256, 128};
    const int uout[3] = {256, 128, 64};
    H = 32;
    for (int i = 0; i < 3; ++i) {
        others(cur, o);
        convrun(cur, ups_[i].up, nullptr, o[0], uin[i], H, uin[i], 4, 2, 1, true);
        H <<= 1;
        resnet(ups_[i].r0, o[0], o[1], o[2], cur, uin[i], uout[i], H);
        resnet(ups_[i].r1, o[1], o[0], o[2], cur, uout[i], uout[i], H);
        cur = o[0];
    }
    // H == 256, 64 channels

    others(cur, o);
    gnrun(cur, dec_norm, o[0], 64, 65536, true);
    convrun(o[0], dec_out_p, nullptr, recon, 64, 256, 3, 3, 1, 1, false);
}
